// TMSwinBlock_19284403159698
// MI455X (gfx1250) — matrix operand order repaired and checked by reading, not compiled, not run on hardware
//
#include <hip/hip_runtime.h>

#define DEVI __device__ __forceinline__

typedef unsigned short u16t;
typedef __attribute__((ext_vector_type(16))) __bf16 bf16x16;
typedef __attribute__((ext_vector_type(8)))  float  f32x8;
typedef __attribute__((ext_vector_type(4)))  unsigned uv4;
typedef __attribute__((ext_vector_type(4)))  unsigned uintv4;
typedef __attribute__((ext_vector_type(8)))  int      intv8;
typedef __attribute__((ext_vector_type(4)))  int      intv4;

// ---------------- problem constants ----------------
#define BB   16
#define HH   56
#define WW   56
#define CC   96
#define LL   (HH*WW)          // 3136
#define NHD  3
#define WSZ  7
#define SSH  3
#define HID  384
#define NCL  128
#define NWIN 1024             // BB * 8 * 8
#define BT   (BB*LL)          // 50176

#if defined(__has_builtin)
#  if __has_builtin(__builtin_amdgcn_tensor_load_to_lds) && __has_builtin(__builtin_amdgcn_s_wait_tensorcnt)
#    define HAVE_TDM 1
#  else
#    define HAVE_TDM 0
#  endif
#else
#  define HAVE_TDM 0
#endif

union FragU { bf16x16 v; u16t u[16]; uv4 q[2]; };

DEVI u16t f2bf(float f){
  unsigned x = __builtin_bit_cast(unsigned, f);
  unsigned r = x + 0x7fffu + ((x >> 16) & 1u);   // round-to-nearest-even
  return (u16t)(r >> 16);
}
DEVI float bf2f(u16t h){
  unsigned x = ((unsigned)h) << 16;
  return __builtin_bit_cast(float, x);
}
DEVI float sigm(float x){ return 1.0f / (1.0f + __expf(-x)); }
DEVI f32x8 zero8(){ f32x8 z = {0.f,0.f,0.f,0.f,0.f,0.f,0.f,0.f}; return z; }

// A fragment (16x32, M x K) from row-major bf16 buffer.
// lane<16: row=lane&15, K = {k0..k0+7, k0+16..k0+23}; lane>=16: K = {k0+8.., k0+24..}
DEVI bf16x16 load_a16(const u16t* a, int lda, int m0, int k0){
  int lane = (int)(threadIdx.x & 31u);
  const u16t* p = a + (size_t)(m0 + (lane & 15)) * lda + k0 + (lane >> 4) * 8;
  FragU f;
  f.q[0] = *(const uv4*)(p);
  f.q[1] = *(const uv4*)(p + 16);
  return f.v;
}
// B fragment (32x16, K x N) where memory holds B^T row-major, i.e. W[n][k]
DEVI bf16x16 load_bT16(const u16t* w, int ldk, int n0, int k0){
  int lane = (int)(threadIdx.x & 31u);
  const u16t* p = w + (size_t)(n0 + (lane & 15)) * ldk + k0 + (lane >> 4) * 8;
  FragU f;
  f.q[0] = *(const uv4*)(p);
  f.q[1] = *(const uv4*)(p + 16);
  return f.v;
}
DEVI f32x8 wmma_bf(bf16x16 a, bf16x16 b, f32x8 c){
  return __builtin_amdgcn_wmma_f32_16x16x32_bf16(false, a, false, b, (short)0, c, false, false);
}

// ---- Tensor Data Mover: DMA `bytes` (multiple of 8) from global -> LDS ----
// D# per CDNA5 ISA: group0 {count=1, lds_addr, global_addr,
// type=2}; group1 {data_size=8B, tensor_dim0=tile_dim0=stride=bytes/8, dims1=1}.
DEVI void tdm_load_to_lds(const void* gptr, void* lds_ptr, unsigned bytes){
#if HAVE_TDM
  unsigned long long ga = (unsigned long long)gptr;
  unsigned lds_off = (unsigned)(unsigned long long)lds_ptr; // low 32b of LDS aperture addr
  unsigned units = bytes >> 3;                              // 8-byte elements
  uintv4 g0;
  g0[0] = 1u;                                               // count=1, user mode
  g0[1] = lds_off;                                          // lds_addr
  g0[2] = (unsigned)(ga & 0xffffffffu);                     // global_addr[31:0]
  g0[3] = (unsigned)((ga >> 32) & 0x1ffffffu) | (2u << 30); // addr[56:32] | type=2
  intv8 g1;
  g1[0] = (int)(3u << 16);                                  // data_size=8B, no flags
  g1[1] = (int)((units & 0xffffu) << 16);                   // tensor_dim0[15:0]
  g1[2] = (int)(((units >> 16) & 0xffffu) | (1u << 16));    // dim0[31:16] | tensor_dim1=1
  g1[3] = (int)((units & 0xffffu) << 16);                   // tile_dim0
  g1[4] = 1;                                                // tile_dim1=1, tile_dim2=0
  g1[5] = (int)units;                                       // tensor_dim0_stride[31:0]
  g1[6] = 0; g1[7] = 0;
  intv4 gz = {0,0,0,0};
#if __clang_major__ >= 23
  intv8 gz8 = {0,0,0,0,0,0,0,0};
  __builtin_amdgcn_tensor_load_to_lds(g0, g1, gz, gz, gz8, 0);
#else
  __builtin_amdgcn_tensor_load_to_lds(g0, g1, gz, gz, 0);
#endif
#else
  // fallback: cooperative wave copy
  const uv4* s = (const uv4*)gptr;
  uv4* d = (uv4*)lds_ptr;
  for (unsigned i = (threadIdx.x & 31u); i < (bytes >> 4); i += 32) d[i] = s[i];
#endif
}
DEVI void tdm_wait(){
#if HAVE_TDM
  __builtin_amdgcn_s_wait_tensorcnt(0);
#endif
}

// ================= K0: weight conversion (fp32 -> bf16, STE mask) ==========
__global__ void k_prep(const float* __restrict__ qkv_w, const float* __restrict__ proj_w,
                       const float* __restrict__ pin_w, const float* __restrict__ tm_inc,
                       const float* __restrict__ tm_out,
                       u16t* qkvw, u16t* projw, u16t* pinw, u16t* mask, u16t* tmoutT){
  int i = blockIdx.x * blockDim.x + threadIdx.x;
  const int n1 = 288*96, n2 = 96*96, n3 = 384*96, n4 = 128*768, n5 = 96*128;
  if (i < n1){ qkvw[i] = f2bf(qkv_w[i]); return; }  i -= n1;
  if (i < n2){ projw[i] = f2bf(proj_w[i]); return; } i -= n2;
  if (i < n3){ pinw[i]  = f2bf(pin_w[i]);  return; } i -= n3;
  if (i < n4){ mask[i]  = f2bf(tm_inc[i] > 0.0f ? 1.0f : 0.0f); return; } i -= n4; // sigmoid(x)>0.5 <=> x>0
  if (i < n5){ int n = i / 128, k = i & 127; tmoutT[i] = f2bf(tm_out[k*96 + n]); }
}

// ================= K1: LN1 + shifted-window gather -> bf16 rows ============
// block = 256 threads = 16 tokens x 16 lanes; grid = NWIN*4 (one row-tile each)
__global__ void k_ln1_win(const float* __restrict__ x, const float* __restrict__ g,
                          const float* __restrict__ b, u16t* __restrict__ ln1){
  int blk = blockIdx.x;
  int w  = blk >> 2, rt = blk & 3;
  int grp = (int)(threadIdx.x >> 4), sub = (int)(threadIdx.x & 15u);
  int t = rt * 16 + grp;
  size_t orow = ((size_t)w * 64 + t) * CC;
  if (t >= 49){
    #pragma unroll
    for (int j = 0; j < 6; ++j) ln1[orow + sub*6 + j] = 0;
    return;
  }
  int bb = w >> 6, wy = (w >> 3) & 7, wx = w & 7;
  int ty = t / WSZ, tx = t - ty * WSZ;
  int h = wy*WSZ + ty + SSH; if (h >= HH) h -= HH;   // roll(-SS): shifted[i] = x[i+SS]
  int c = wx*WSZ + tx + SSH; if (c >= WW) c -= WW;
  size_t irow = ((size_t)bb * LL + h * WW + c) * CC;
  float vals[6]; float s = 0.f, s2 = 0.f;
  #pragma unroll
  for (int j = 0; j < 6; ++j){ float v = x[irow + sub*6 + j]; vals[j] = v; s += v; s2 += v*v; }
  #pragma unroll
  for (int m = 1; m < 16; m <<= 1){ s += __shfl_xor(s, m, 16); s2 += __shfl_xor(s2, m, 16); }
  float mean = s * (1.f/CC);
  float var  = s2 * (1.f/CC) - mean*mean;
  float inv  = rsqrtf(var + 1e-5f);
  #pragma unroll
  for (int j = 0; j < 6; ++j){
    int cc = sub*6 + j;
    ln1[orow + cc] = f2bf((vals[j] - mean) * inv * g[cc] + b[cc]);
  }
}

// ================= K2: QKV GEMM (64x96 per block, K=96) ====================
// grid (NWIN, 3=q/k/v); 8 waves; weights staged in LDS by the Tensor Data Mover
__global__ void k_qkv(const u16t* __restrict__ ln1, const u16t* __restrict__ qkvw,
                      const float* __restrict__ qkv_b,
                      u16t* __restrict__ qb, u16t* __restrict__ kb, u16t* __restrict__ vT){
  __shared__ u16t Bs[288 * 96];                    // 55296 B
  int w = blockIdx.x, which = blockIdx.y;
  int wv = (int)(threadIdx.x >> 5), lane = (int)(threadIdx.x & 31u);
  if (wv == 0){
    tdm_load_to_lds(qkvw, &Bs[0], 288 * 96 * 2);   // TDM DMA: whole W_qkv -> LDS
    tdm_wait();                                    // s_wait_tensorcnt 0
  }
  __syncthreads();

  int rt = wv >> 1, ct0 = (wv & 1) * 3;
  const u16t* A = ln1 + (size_t)w * 64 * CC;
  f32x8 acc[3] = { zero8(), zero8(), zero8() };
  #pragma unroll
  for (int ks = 0; ks < 3; ++ks){
    bf16x16 a = load_a16(A, CC, rt*16, ks*32);
    #pragma unroll
    for (int j = 0; j < 3; ++j){
      bf16x16 bf = load_bT16(&Bs[0], CC, which*96 + (ct0 + j)*16, ks*32);
      acc[j] = wmma_bf(a, bf, acc[j]);
    }
  }
  int kh = lane >> 4, ln = lane & 15;
  #pragma unroll
  for (int j = 0; j < 3; ++j){
    int nl = (ct0 + j)*16 + ln;          // 0..95 within q/k/v
    int ng = which*96 + nl;              // 0..287
    float bias = qkv_b[ng];
    int head = nl >> 5, d = nl & 31;
    size_t wh = (size_t)w * NHD + head;
    #pragma unroll
    for (int i = 0; i < 8; ++i){
      int t = rt*16 + kh*8 + i;
      u16t hv = f2bf(acc[j][i] + bias);
      if      (which == 0) qb[(wh*64 + t)*32 + d] = hv;
      else if (which == 1) kb[(wh*64 + t)*32 + d] = hv;
      else                 vT[(wh*32 + d)*64 + t] = hv;   // store v transposed
    }
  }
}

// ================= K3: window attention (per window, head) =================
// block = 128 threads (4 waves, wave = S row-tile); S=qk^T, softmax, O=Pv
__global__ void k_attn(const u16t* __restrict__ qb, const u16t* __restrict__ kb,
                       const u16t* __restrict__ vT, u16t* __restrict__ attnA){
  __shared__ u16t P[64 * 64];
  int w = blockIdx.x, hh = blockIdx.y;
  int wv = (int)(threadIdx.x >> 5), lane = (int)(threadIdx.x & 31u);
  int rt = wv, kh = lane >> 4, ln = lane & 15;
  size_t wh = (size_t)w * NHD + hh;
  const u16t* q  = qb + wh * 64 * 32;
  const u16t* k  = kb + wh * 64 * 32;
  const u16t* v  = vT + wh * 32 * 64;

  bf16x16 a = load_a16(q, 32, rt*16, 0);
  f32x8 s[4];
  #pragma unroll
  for (int ct = 0; ct < 4; ++ct)
    s[ct] = wmma_bf(a, load_bT16(k, 32, ct*16, 0), zero8());

  const float scale = 0.17677669529663687f;   // 1/sqrt(32)
  #pragma unroll
  for (int i = 0; i < 8; ++i){
    int m = rt*16 + kh*8 + i;
    float e[4]; float mx = -1e30f;
    #pragma unroll
    for (int ct = 0; ct < 4; ++ct){
      int n = ct*16 + ln;
      float vv = (n < 49) ? s[ct][i] * scale : -1e30f;
      e[ct] = vv; mx = fmaxf(mx, vv);
    }
    #pragma unroll
    for (int d = 1; d < 16; d <<= 1) mx = fmaxf(mx, __shfl_xor(mx, d, 16));
    float sum = 0.f;
    #pragma unroll
    for (int ct = 0; ct < 4; ++ct){ e[ct] = __expf(e[ct] - mx); sum += e[ct]; }
    #pragma unroll
    for (int d = 1; d < 16; d <<= 1) sum += __shfl_xor(sum, d, 16);
    float inv = 1.f / sum;
    #pragma unroll
    for (int ct = 0; ct < 4; ++ct) P[m*64 + ct*16 + ln] = f2bf(e[ct] * inv);
  }
  __syncthreads();

  f32x8 o[2] = { zero8(), zero8() };
  #pragma unroll
  for (int ks = 0; ks < 2; ++ks){
    bf16x16 ap = load_a16(&P[0], 64, rt*16, ks*32);
    #pragma unroll
    for (int ct = 0; ct < 2; ++ct)
      o[ct] = wmma_bf(ap, load_bT16(v, 64, ct*16, ks*32), o[ct]);
  }
  #pragma unroll
  for (int ct = 0; ct < 2; ++ct)
    #pragma unroll
    for (int i = 0; i < 8; ++i){
      int m = rt*16 + kh*8 + i;
      attnA[((size_t)w*64 + m)*CC + hh*32 + ct*16 + ln] = f2bf(o[ct][i]);
    }
}

// ================= K4: proj GEMM + un-shift scatter + residual =============
__global__ void k_proj(const u16t* __restrict__ attnA, const u16t* __restrict__ projw,
                       const float* __restrict__ proj_b, const float* __restrict__ x,
                       float* __restrict__ x1){
  int w = blockIdx.x;
  int wv = (int)(threadIdx.x >> 5), lane = (int)(threadIdx.x & 31u);
  int rt = wv >> 1, ct0 = (wv & 1) * 3;
  const u16t* A = attnA + (size_t)w * 64 * CC;
  f32x8 acc[3] = { zero8(), zero8(), zero8() };
  #pragma unroll
  for (int ks = 0; ks < 3; ++ks){
    bf16x16 a = load_a16(A, CC, rt*16, ks*32);
    #pragma unroll
    for (int j = 0; j < 3; ++j)
      acc[j] = wmma_bf(a, load_bT16(projw, CC, (ct0 + j)*16, ks*32), acc[j]);
  }
  int bb = w >> 6, wy = (w >> 3) & 7, wx = w & 7;
  int kh = lane >> 4, ln = lane & 15;
  #pragma unroll
  for (int j = 0; j < 3; ++j){
    int n = (ct0 + j)*16 + ln;
    float bias = proj_b[n];
    #pragma unroll
    for (int i = 0; i < 8; ++i){
      int t = rt*16 + kh*8 + i;
      if (t < 49){
        int ty = t / WSZ, tx = t - ty * WSZ;
        int h = wy*WSZ + ty + SSH; if (h >= HH) h -= HH;   // roll back (+SS)
        int c = wx*WSZ + tx + SSH; if (c >= WW) c -= WW;
        size_t row = (size_t)bb * LL + h * WW + c;
        x1[row*CC + n] = x[row*CC + n] + acc[j][i] + bias;
      }
    }
  }
}

// ================= K5a: double LayerNorm (norm2 then ffn_norm) =============
__global__ void k_ln2(const float* __restrict__ x1, const float* __restrict__ g2,
                      const float* __restrict__ b2, const float* __restrict__ gf,
                      const float* __restrict__ bf_, u16t* __restrict__ ln2){
  size_t row = (size_t)blockIdx.x * 16 + (threadIdx.x >> 4);
  int sub = (int)(threadIdx.x & 15u);
  const float* xr = x1 + row * CC;
  float vals[6]; float s = 0.f, s2 = 0.f;
  #pragma unroll
  for (int j = 0; j < 6; ++j){ float v = xr[sub*6 + j]; vals[j] = v; s += v; s2 += v*v; }
  #pragma unroll
  for (int m = 1; m < 16; m <<= 1){ s += __shfl_xor(s, m, 16); s2 += __shfl_xor(s2, m, 16); }
  float mean = s * (1.f/CC), var = s2 * (1.f/CC) - mean*mean;
  float inv = rsqrtf(var + 1e-5f);
  float zs = 0.f, zs2 = 0.f;
  #pragma unroll
  for (int j = 0; j < 6; ++j){
    int cc = sub*6 + j;
    float z = (vals[j] - mean) * inv * g2[cc] + b2[cc];
    vals[j] = z; zs += z; zs2 += z*z;
  }
  #pragma unroll
  for (int m = 1; m < 16; m <<= 1){ zs += __shfl_xor(zs, m, 16); zs2 += __shfl_xor(zs2, m, 16); }
  float m2 = zs * (1.f/CC), v2 = zs2 * (1.f/CC) - m2*m2;
  float inv2 = rsqrtf(v2 + 1e-5f);
  #pragma unroll
  for (int j = 0; j < 6; ++j){
    int cc = sub*6 + j;
    ln2[row*CC + cc] = f2bf((vals[j] - m2) * inv2 * gf[cc] + bf_[cc]);
  }
}

// ================= K5b: pin GEMM + sigmoid -> ybuf =========================
// grid (BT/64, 4); 64 rows x 96 cols per block, K=96
__global__ void k_pin(const u16t* __restrict__ ln2, const u16t* __restrict__ pinw,
                      const float* __restrict__ pin_b, u16t* __restrict__ ybuf){
  int rb = blockIdx.x, nb = blockIdx.y;
  int wv = (int)(threadIdx.x >> 5), lane = (int)(threadIdx.x & 31u);
  int rt = wv >> 1, ct0 = (wv & 1) * 3;
  const u16t* A = ln2 + (size_t)rb * 64 * CC;
  f32x8 acc[3] = { zero8(), zero8(), zero8() };
  #pragma unroll
  for (int ks = 0; ks < 3; ++ks){
    bf16x16 a = load_a16(A, CC, rt*16, ks*32);
    #pragma unroll
    for (int j = 0; j < 3; ++j)
      acc[j] = wmma_bf(a, load_bT16(pinw, CC, nb*96 + (ct0 + j)*16, ks*32), acc[j]);
  }
  int kh = lane >> 4, ln = lane & 15;
  #pragma unroll
  for (int j = 0; j < 3; ++j){
    int n = nb*96 + (ct0 + j)*16 + ln;
    float bias = pin_b[n];
    #pragma unroll
    for (int i = 0; i < 8; ++i){
      int m = rt*16 + kh*8 + i;
      ybuf[((size_t)rb*64 + m)*HID + n] = f2bf(sigm(acc[j][i] + bias));
    }
  }
}

// ================= K6: fused TM clauses + output + gate + residual =========
// grid BT/64; stage1: 64x128 clause GEMM (K=768, A = log-literals on the fly)
// stage2: 64x96 via LDS-staged clauses; tm_out staged by TDM overlapping stage1
__global__ void k_tm(const u16t* __restrict__ ybuf, const u16t* __restrict__ mask,
                     const u16t* __restrict__ tmoutT, const float* __restrict__ x1,
                     const float* __restrict__ gatep, float* __restrict__ out){
  __shared__ u16t TOUT[96 * 128];                 // 24576 B, filled by TDM
  __shared__ u16t CL[64 * 128];
  int rb = blockIdx.x;
  int wv = (int)(threadIdx.x >> 5), lane = (int)(threadIdx.x & 31u);
  int kh = lane >> 4, ln = lane & 15;
  if (wv == 0)
    tdm_load_to_lds(tmoutT, &TOUT[0], 96 * 128 * 2);  // async; overlaps stage 1

  // ---- stage 1: clauses = exp( log(lits) @ mask^T ) ----
  int rt = wv >> 1, ct0 = (wv & 1) * 4;
  f32x8 acc[4] = { zero8(), zero8(), zero8(), zero8() };
  int m = rt*16 + ln;
  const u16t* yr = ybuf + ((size_t)rb*64 + m) * HID;
  for (int kt = 0; kt < 24; ++kt){
    int k0 = kt * 32;
    __builtin_prefetch(mask + (size_t)(ct0*16)*768 + k0 + 32, 0, 2);
    FragU f;
    #pragma unroll
    for (int j = 0; j < 16; ++j){
      int kk = k0 + kh*8 + ((j < 8) ? j : (8 + j));   // A-frag K mapping
      float lit = (kk < HID) ? bf2f(yr[kk]) : (1.0f - bf2f(yr[kk - HID]));
      lit = fminf(fmaxf(lit, 1e-6f), 1.0f);
      f.u[j] = f2bf(__logf(lit));
    }
    #pragma unroll
    for (int j = 0; j < 4; ++j)
      acc[j] = wmma_bf(f.v, load_bT16(mask, 768, (ct0 + j)*16, k0), acc[j]);
  }
  #pragma unroll
  for (int j = 0; j < 4; ++j)
    #pragma unroll
    for (int i = 0; i < 8; ++i){
      int mm = rt*16 + kh*8 + i;
      CL[mm*128 + (ct0 + j)*16 + ln] = f2bf(__expf(acc[j][i]));
    }
  if (wv == 0) tdm_wait();                        // s_wait_tensorcnt 0
  __syncthreads();

  // ---- stage 2: logits = clauses @ tm_out ; gate + residual ----
  int cs0 = (wv & 1) * 3;
  f32x8 o[3] = { zero8(), zero8(), zero8() };
  #pragma unroll
  for (int ks = 0; ks < 4; ++ks){
    bf16x16 a = load_a16(&CL[0], 128, rt*16, ks*32);
    #pragma unroll
    for (int j = 0; j < 3; ++j)
      o[j] = wmma_bf(a, load_bT16(&TOUT[0], 128, (cs0 + j)*16, ks*32), o[j]);
  }
  float g = sigm(gatep[0]);
  #pragma unroll
  for (int j = 0; j < 3; ++j)
    #pragma unroll
    for (int i = 0; i < 8; ++i){
      int mm = rt*16 + kh*8 + i;
      int n = (cs0 + j)*16 + ln;
      size_t row = (size_t)rb*64 + mm;
      float lg = o[j][i];
      out[row*CC + n] = x1[row*CC + n] + g*lg + (1.f - g)*sigm(lg);
    }
}

// ============================ launcher =====================================
extern "C" void kernel_launch(void* const* d_in, const int* in_sizes, int n_in,
                              void* d_out, int out_size, void* d_ws, size_t ws_size,
                              hipStream_t stream) {
  const float* x       = (const float*)d_in[0];
  const float* n1g     = (const float*)d_in[1];
  const float* n1b     = (const float*)d_in[2];
  const float* qkv_w   = (const float*)d_in[3];
  const float* qkv_b   = (const float*)d_in[4];
  const float* proj_w  = (const float*)d_in[5];
  const float* proj_b  = (const float*)d_in[6];
  const float* n2g     = (const float*)d_in[7];
  const float* n2b     = (const float*)d_in[8];
  const float* fng     = (const float*)d_in[9];
  const float* fnb     = (const float*)d_in[10];
  const float* pin_w   = (const float*)d_in[11];
  const float* pin_b   = (const float*)d_in[12];
  const float* tm_inc  = (const float*)d_in[13];
  const float* tm_out  = (const float*)d_in[14];
  const float* gate    = (const float*)d_in[15];
  float* outp          = (float*)d_out;

  char* ws = (char*)d_ws;
  size_t off = 0;
  auto alloc = [&](size_t sz){ size_t r = off; off = (off + sz + 255) & ~(size_t)255; return r; };
  size_t o_qkvw  = alloc(288*96*2);
  size_t o_projw = alloc(96*96*2);
  size_t o_pinw  = alloc(384*96*2);
  size_t o_mask  = alloc((size_t)128*768*2);
  size_t o_tmout = alloc(96*128*2);
  size_t o_ln1   = alloc((size_t)NWIN*64*96*2);   // also aliased by attnA and ybuf
  size_t o_q     = alloc((size_t)NWIN*NHD*64*32*2);
  size_t o_k     = alloc((size_t)NWIN*NHD*64*32*2);
  size_t o_vT    = alloc((size_t)NWIN*NHD*32*64*2);
  size_t o_x1    = alloc((size_t)BT*96*4);
  size_t o_ln2   = alloc((size_t)BT*96*2);
  (void)ws_size; (void)n_in; (void)in_sizes; (void)out_size;

  u16t* qkvw   = (u16t*)(ws + o_qkvw);
  u16t* projw  = (u16t*)(ws + o_projw);
  u16t* pinw   = (u16t*)(ws + o_pinw);
  u16t* maskb  = (u16t*)(ws + o_mask);
  u16t* tmoutT = (u16t*)(ws + o_tmout);
  u16t* ln1    = (u16t*)(ws + o_ln1);
  u16t* attnA  = (u16t*)(ws + o_ln1);   // alias: ln1 dead after K2
  u16t* ybuf   = (u16t*)(ws + o_ln1);   // alias: spans ln1+q+k and part of vT, dead by then
  u16t* qb     = (u16t*)(ws + o_q);
  u16t* kb     = (u16t*)(ws + o_k);
  u16t* vT     = (u16t*)(ws + o_vT);
  float* x1    = (float*)(ws + o_x1);
  u16t* ln2    = (u16t*)(ws + o_ln2);

  // K0: weights -> bf16 (+ STE mask, tm_out transpose)
  k_prep<<<720, 256, 0, stream>>>(qkv_w, proj_w, pin_w, tm_inc, tm_out,
                                  qkvw, projw, pinw, maskb, tmoutT);
  // K1: LN1 + shifted window gather
  k_ln1_win<<<NWIN*4, 256, 0, stream>>>(x, n1g, n1b, ln1);
  // K2: QKV GEMM (W_qkv staged via Tensor Data Mover)
  k_qkv<<<dim3(NWIN, 3), 256, 0, stream>>>(ln1, qkvw, qkv_b, qb, kb, vT);
  // K3: windowed attention
  k_attn<<<dim3(NWIN, NHD), 128, 0, stream>>>(qb, kb, vT, attnA);
  // K4: proj + un-shift scatter + residual -> x1
  k_proj<<<NWIN, 256, 0, stream>>>(attnA, projw, proj_b, x, x1);
  // K5a: LN(norm2) then LN(ffn_norm)
  k_ln2<<<BT/16, 256, 0, stream>>>(x1, n2g, n2b, fng, fnb, ln2);
  // K5b: pin GEMM + sigmoid
  k_pin<<<dim3(BT/64, 4), 256, 0, stream>>>(ln2, pinw, pin_b, ybuf);
  // K6: fused TM clauses + output + gate + residual (tm_out via TDM)
  k_tm<<<BT/64, 256, 0, stream>>>(ybuf, maskb, tmoutT, x1, gate, outp);
}
